// AttentionLayerTanh_62706522521853
// MI455X (gfx1250) — compile-verified, matrix operand order checked by reading, not run on hardware
//
#include <hip/hip_runtime.h>

// CDNA5 / gfx1250, wave32. Fused tanh-attention layer:
//   one workgroup (256 thr = 8 waves) per batch element.
//   wk GEMM via v_wmma_f32_16x16x32_bf16 from LDS-staged bf16 tiles.
//   value tile prefetched (global_prefetch_b8) at entry to hide the epilogue read.

typedef __attribute__((ext_vector_type(16))) __bf16 v16bf;
typedef __attribute__((ext_vector_type(8)))  __bf16 v8bf;
typedef __attribute__((ext_vector_type(8)))  float  v8f;

#define NTOK   121      // tokens per batch (11x11 grid)
#define DKEY   128      // key feature dim
#define DQ     256      // query dim
#define KFEAT  128      // attention feature dim
#define MPAD   128      // token dim padded to 8 M-tiles
#define KG     160      // GEMM inner dim: 130 padded to 5 chunks of 32
#define LDSLD  160      // LDS row stride (elements) for both tiles

#define SA_BYTES   (MPAD * LDSLD * 2)          // 40960
#define SW_BYTES   (KFEAT * LDSLD * 2)         // 40960
#define SMEM_BYTES (SA_BYTES + SW_BYTES + (256 + 256 + 128 + 128 + 128) * 4)  // 85504

__device__ __forceinline__ unsigned short f2bf(float f) {   // RNE float->bf16
    union { float f; unsigned u; } v; v.f = f;
    unsigned r = v.u + 0x7fffu + ((v.u >> 16) & 1u);
    return (unsigned short)(r >> 16);
}
__device__ __forceinline__ float coordY(int n) { return 0.2f * (float)(n / 11) - 1.0f; }
__device__ __forceinline__ float coordX(int n) { return 0.2f * (float)(n % 11) - 1.0f; }

__global__ __launch_bounds__(256) void attn_tanh_fused(
    const float* __restrict__ query,   // [B,256]
    const float* __restrict__ key,     // [B,121,128]
    const float* __restrict__ value,   // [B,121,128]
    const float* __restrict__ to_u_w,  // [256,128]
    const float* __restrict__ to_u_b,  // [128]
    const float* __restrict__ to_w_w,  // [130,128]
    const float* __restrict__ to_w_b,  // [128]
    const float* __restrict__ to_v_w,  // [128,1]
    const float* __restrict__ to_v_b,  // [1]
    float* __restrict__ out,           // [B,130]
    float* __restrict__ weights)       // [B,121]
{
    extern __shared__ unsigned char smem[];
    unsigned short* sA  = (unsigned short*)smem;                // [128][160] bf16: key_pe, zero-padded
    unsigned short* sWt = (unsigned short*)(smem + SA_BYTES);   // [128][160] bf16: to_w_w transposed
    float* sQ  = (float*)(smem + SA_BYTES + SW_BYTES);          // 256
    float* sUp = sQ  + 256;                                     // 256 (u partials)
    float* sU  = sUp + 256;                                     // 128 (u + biases)
    float* sV  = sU  + 128;                                     // 128 (to_v_w)
    float* sS  = sV  + 128;                                     // 128 (scores -> weights)

    const int tid = threadIdx.x;
    const int b   = blockIdx.x;

    // ---- prefetch this batch's value tile (62 KB) while we do the GEMM ----
    // lowers to global_prefetch_b8; one lane per 256B chunk pulls a cacheline.
    {
        const char* vbp = (const char*)(value + (size_t)b * (NTOK * DKEY));
        for (int i = tid; i < (NTOK * DKEY * 4) / 256; i += 256)
            __builtin_prefetch(vbp + i * 256, 0, 1);
    }

    // ---- zero both bf16 tiles (covers K-padding and M-padding rows) ----
    {
        unsigned* zp = (unsigned*)smem;                         // 81920 B = 20480 dwords
        for (int i = tid; i < (SA_BYTES + SW_BYTES) / 4; i += 256) zp[i] = 0u;
    }
    __syncthreads();

    // ---- stage key -> sA (f32 -> bf16), cols 0..127 ----
    const float* kb = key + (size_t)b * (NTOK * DKEY);
    for (int i = tid; i < NTOK * (DKEY / 4); i += 256) {
        int row = i >> 5, q = i & 31;                           // 32 float4 per row
        float4 f = *(const float4*)(kb + row * DKEY + q * 4);
        unsigned lo = (unsigned)f2bf(f.x) | ((unsigned)f2bf(f.y) << 16);
        unsigned hi = (unsigned)f2bf(f.z) | ((unsigned)f2bf(f.w) << 16);
        unsigned* dst = (unsigned*)(sA + row * LDSLD + q * 4);
        dst[0] = lo; dst[1] = hi;
    }
    // positional coords -> cols 128,129
    for (int i = tid; i < NTOK * 2; i += 256) {
        int row = i >> 1, cc = i & 1;
        sA[row * LDSLD + DKEY + cc] = f2bf(cc ? coordX(row) : coordY(row));
    }
    // ---- stage to_w_w transposed: sWt[n][k] = W[k][n] (coalesced global reads) ----
    for (int i = tid; i < 130 * KFEAT; i += 256) {
        int n = i & 127, k = i >> 7;
        sWt[n * LDSLD + k] = f2bf(to_w_w[k * KFEAT + n]);
    }
    sQ[tid] = query[(size_t)b * DQ + tid];
    if (tid < KFEAT) sV[tid] = to_v_w[tid];
    __syncthreads();

    // ---- u = q @ to_u_w + to_u_b + to_w_b (VALU; split dot over 2 halves) ----
    {
        int k = tid & 127, half = tid >> 7;
        const float* wp = to_u_w + (size_t)half * 128 * KFEAT + k;
        const float* qp = sQ + half * 128;
        float acc = 0.f;
        #pragma unroll 8
        for (int d = 0; d < 128; ++d) acc += qp[d] * wp[d * KFEAT];
        sUp[tid] = acc;
    }
    __syncthreads();
    if (tid < KFEAT) sU[tid] = sUp[tid] + sUp[tid + 128] + to_u_b[tid] + to_w_b[tid];
    __syncthreads();

    // ---- GEMM: wk[128x128] = key_pe[128x160] @ Wt^T, bf16 WMMA, f32 accum ----
    const int wave = tid >> 5, lane = tid & 31;
    const int c = lane & 15, hs = lane >> 4;    // lane layout: c = row(A)/col(B), hs = K-half select
    const int m0 = wave * 16;                   // each wave: 16 rows x 128 cols

    v8f acc[8];
    #pragma unroll
    for (int t = 0; t < 8; ++t) acc[t] = (v8f){0.f,0.f,0.f,0.f,0.f,0.f,0.f,0.f};

    const unsigned short* aBase = sA + (m0 + c) * LDSLD + hs * 8;
    #pragma unroll
    for (int k0 = 0; k0 < KG; k0 += 32) {
        // load A fragment + ALL 8 B fragments first, then issue 8 WMMAs:
        // lets the compiler batch ds_load_b128s and overlap them with matrix math.
        v8bf alo = *(const v8bf*)(aBase + k0);          // K = k0 + hs*8 .. +7
        v8bf ahi = *(const v8bf*)(aBase + k0 + 16);     // K = k0 + 16 + hs*8 .. +7
        v16bf afrag = __builtin_shufflevector(alo, ahi,
            0,1,2,3,4,5,6,7,8,9,10,11,12,13,14,15);
        v16bf bfrag[8];
        #pragma unroll
        for (int t = 0; t < 8; ++t) {
            const unsigned short* bBase = sWt + (t * 16 + c) * LDSLD + hs * 8 + k0;
            v8bf blo = *(const v8bf*)(bBase);
            v8bf bhi = *(const v8bf*)(bBase + 16);
            bfrag[t] = __builtin_shufflevector(blo, bhi,
                0,1,2,3,4,5,6,7,8,9,10,11,12,13,14,15);
        }
        #pragma unroll
        for (int t = 0; t < 8; ++t)
            acc[t] = __builtin_amdgcn_wmma_f32_16x16x32_bf16(
                false, afrag, false, bfrag[t], (short)0, acc[t], false, false);
    }

    // ---- epilogue on C-fragment layout: score[row] = sum_col tanh(wk+u)*v ----
    float partial[8];
    #pragma unroll
    for (int r = 0; r < 8; ++r) partial[r] = 0.f;
    #pragma unroll
    for (int t = 0; t < 8; ++t) {
        int col = t * 16 + c;
        float uv = sU[col], vv = sV[col];
        #pragma unroll
        for (int r = 0; r < 8; ++r)
            partial[r] += tanhf(acc[t][r] + uv) * vv;
    }
    #pragma unroll
    for (int r = 0; r < 8; ++r) {                     // reduce 16 lanes sharing a row
        float p = partial[r];
        p += __shfl_xor(p, 1, 32);
        p += __shfl_xor(p, 2, 32);
        p += __shfl_xor(p, 4, 32);
        p += __shfl_xor(p, 8, 32);
        partial[r] = p;
    }
    if (c == 0) {                                     // lanes 0 (rows m0..+7) and 16 (rows m0+8..+15)
        float vb0 = to_v_b[0];
        #pragma unroll
        for (int r = 0; r < 8; ++r) {
            int row = m0 + hs * 8 + r;
            if (row < NTOK) sS[row] = partial[r] + vb0;
        }
    }
    __syncthreads();

    // ---- softmax over 121 scores (wave 0) ----
    if (wave == 0) {
        float v0[4]; float mx = -3.0e38f;
        #pragma unroll
        for (int j = 0; j < 4; ++j) {
            int n = lane + j * 32;
            v0[j] = (n < NTOK) ? sS[n] : -3.0e38f;
            mx = fmaxf(mx, v0[j]);
        }
        for (int m = 1; m < 32; m <<= 1) mx = fmaxf(mx, __shfl_xor(mx, m, 32));
        float sm = 0.f;
        #pragma unroll
        for (int j = 0; j < 4; ++j) { v0[j] = __expf(v0[j] - mx); sm += v0[j]; }
        for (int m = 1; m < 32; m <<= 1) sm += __shfl_xor(sm, m, 32);
        float inv = 1.0f / sm;
        #pragma unroll
        for (int j = 0; j < 4; ++j) {
            int n = lane + j * 32;
            if (n < NTOK) {
                float w = v0[j] * inv;
                sS[n] = w;
                weights[(size_t)b * NTOK + n] = w;
            }
        }
    }
    __syncthreads();

    // ---- output[130] = weights @ value_pe (coalesced: lane d reads value[:,d]) ----
    if (tid < DKEY + 2) {
        const float* vb = value + (size_t)b * (NTOK * DKEY);
        float a = 0.f;
        if (tid < DKEY) {
            #pragma unroll 4
            for (int n = 0; n < NTOK; ++n) a += sS[n] * vb[n * DKEY + tid];
        } else if (tid == DKEY) {
            for (int n = 0; n < NTOK; ++n) a += sS[n] * coordY(n);
        } else {
            for (int n = 0; n < NTOK; ++n) a += sS[n] * coordX(n);
        }
        out[(size_t)b * (DKEY + 2) + tid] = a;
    }
}

extern "C" void kernel_launch(void* const* d_in, const int* in_sizes, int n_in,
                              void* d_out, int out_size, void* d_ws, size_t ws_size,
                              hipStream_t stream) {
    const float* query  = (const float*)d_in[0];
    const float* key    = (const float*)d_in[1];
    const float* value  = (const float*)d_in[2];
    const float* to_u_w = (const float*)d_in[3];
    const float* to_u_b = (const float*)d_in[4];
    const float* to_w_w = (const float*)d_in[5];
    const float* to_w_b = (const float*)d_in[6];
    const float* to_v_w = (const float*)d_in[7];
    const float* to_v_b = (const float*)d_in[8];

    const int B = in_sizes[0] / DQ;            // 8192
    float* out     = (float*)d_out;            // [B,130] then [B,121], flat
    float* weights = out + (size_t)B * (DKEY + 2);

    attn_tanh_fused<<<dim3(B), dim3(256), SMEM_BYTES, stream>>>(
        query, key, value, to_u_w, to_u_b, to_w_w, to_w_b, to_v_w, to_v_b,
        out, weights);
}
